// SABlock_20220706030151
// MI455X (gfx1250) — matrix operand order repaired and checked by reading, not compiled, not run on hardware
//
#include <hip/hip_runtime.h>
#include <hip/hip_bf16.h>

#define DIM    512
#define HEADS  8
#define DHEAD  64
#define BB     4
#define NN     2048
#define M_TOTAL (BB*NN)   // 8192 rows
#define KT     64         // attention key-tile

typedef __attribute__((ext_vector_type(16))) _Float16 v16h;
typedef __attribute__((ext_vector_type(8)))  float    v8f;
typedef __attribute__((ext_vector_type(4)))  unsigned u32x4;
typedef __attribute__((ext_vector_type(8)))  int      i32x8;
typedef __attribute__((ext_vector_type(4)))  int      i32x4;

// ---------------------------------------------------------------------------
// WMMA fragment helpers (wave32; A/B use contiguous k = 16*(lane/16)+s, not the ISA table order)
// A 16x32 f16: lane = M (mod 16), K striped over 8 VGPRs, +16 for lanes 16-31
// B 32x16 f16: lane = N (mod 16), K striped over 8 VGPRs, +16 for lanes 16-31
// C/D 16x16 f32: lane = N (mod 16), M = vgpr + 8*(lane/16)
// ---------------------------------------------------------------------------
__device__ __forceinline__ int kpatA(int p, int g) {
  return 2 * p + 16 * g;
}
__device__ __forceinline__ int kpatB(int p, int g) {
  return 2 * p + 16 * g;
}

__device__ __forceinline__ v16h load_a(const _Float16* ptr, int ld) {
  int lane = threadIdx.x & 31, m = lane & 15, g = lane >> 4;
  v16h r;
#pragma unroll
  for (int p = 0; p < 8; ++p) {
    int k = kpatA(p, g);
    r[2 * p]     = ptr[m * ld + k];
    r[2 * p + 1] = ptr[m * ld + k + 1];
  }
  return r;
}
// B stored K-major (row-major KxN)
__device__ __forceinline__ v16h load_b(const _Float16* ptr, int ld) {
  int lane = threadIdx.x & 31, n = lane & 15, g = lane >> 4;
  v16h r;
#pragma unroll
  for (int p = 0; p < 8; ++p) {
    int k = kpatB(p, g);
    r[2 * p]     = ptr[k * ld + n];
    r[2 * p + 1] = ptr[(k + 1) * ld + n];
  }
  return r;
}
// B fragment from N-major storage (mem[n][k]) -- e.g. K matrix for Q*K^T
__device__ __forceinline__ v16h load_bt(const _Float16* ptr, int ld) {
  int lane = threadIdx.x & 31, n = lane & 15, g = lane >> 4;
  v16h r;
#pragma unroll
  for (int p = 0; p < 8; ++p) {
    int k = kpatB(p, g);
    r[2 * p]     = ptr[n * ld + k];
    r[2 * p + 1] = ptr[n * ld + k + 1];
  }
  return r;
}
__device__ __forceinline__ v8f wmma_f16(v16h a, v16h b, v8f c) {
  return __builtin_amdgcn_wmma_f32_16x16x32_f16(false, a, false, b, (short)0, c,
                                                false, false);
}

// ---------------------------------------------------------------------------
// Row (16-lane) max-allreduce on the VALU via DPP16 row_ror rotations.
// ---------------------------------------------------------------------------
template <int CTRL>
__device__ __forceinline__ float max_ror(float v) {
  int s = __builtin_amdgcn_update_dpp(0, __builtin_bit_cast(int, v), CTRL,
                                      0xf, 0xf, true);
  return fmaxf(v, __builtin_bit_cast(float, s));
}
__device__ __forceinline__ float rowmax16(float v) {
  v = max_ror<0x121>(v);  // row_ror:1
  v = max_ror<0x122>(v);  // row_ror:2
  v = max_ror<0x124>(v);  // row_ror:4
  v = max_ror<0x128>(v);  // row_ror:8
  return v;
}

// ---------------------------------------------------------------------------
// Tensor Data Mover: DMA a 64x64 f16 tile (row stride 64 elems) into LDS.
// D# per ISA 8.3/8.4: group0 = {count=1, lds_addr, global_addr, type=2},
// group1 = {data_size=2B, tensor_dim0/1=64, tile_dim0/1=64, dim0_stride=64}.
// Tracked by TENSORcnt; issued by one wave, published via barrier.
// ---------------------------------------------------------------------------
__device__ __forceinline__ void tdm_load_64x64(_Float16* lds_dst,
                                               const _Float16* gsrc) {
  const unsigned long long ga = (unsigned long long)gsrc;
  const unsigned lds_addr = (unsigned)(unsigned long long)lds_dst;
  u32x4 g0 = {1u,                                   // count=1 (valid user D#)
              lds_addr,                             // lds_addr [63:32]
              (unsigned)ga,                         // global_addr lo
              (unsigned)(ga >> 32) | (2u << 30)};   // addr hi | type=2
  i32x8 g1 = {(int)0x00010000,                      // data_size=1 (2 bytes)
              (int)(64u << 16),                     // tensor_dim0 = 64 (lo16)
              (int)(64u << 16),                     // tensor_dim1 = 64 (lo16)
              (int)(64u << 16),                     // tile_dim0 = 64
              (int)64,                              // tile_dim1 = 64
              (int)64,                              // tensor_dim0_stride = 64
              0, 0};
  i32x4 g2 = {0, 0, 0, 0};
  i32x4 g3 = {0, 0, 0, 0};
#if __clang_major__ >= 23
  i32x8 g4 = {0, 0, 0, 0, 0, 0, 0, 0};
  __builtin_amdgcn_tensor_load_to_lds(g0, g1, g2, g3, g4, 0);
#else
  __builtin_amdgcn_tensor_load_to_lds(g0, g1, g2, g3, 0);
#endif
}

// ---------------------------------------------------------------------------
// LayerNorm + positional embedding -> f16
// ---------------------------------------------------------------------------
__global__ __launch_bounds__(256) void ln_pe(const float* __restrict__ x,
                                             const float* __restrict__ w,
                                             const float* __restrict__ b,
                                             const float* __restrict__ pe,
                                             _Float16* __restrict__ h16) {
  __shared__ float sA[256], sB[256];
  const int row = blockIdx.x, t = threadIdx.x;
  const float* xr = x + (size_t)row * DIM;
  float v0 = xr[t], v1 = xr[t + 256];
  sA[t] = v0 + v1;
  sB[t] = v0 * v0 + v1 * v1;
  __syncthreads();
#pragma unroll
  for (int off = 128; off > 0; off >>= 1) {
    if (t < off) { sA[t] += sA[t + off]; sB[t] += sB[t + off]; }
    __syncthreads();
  }
  float mu   = sA[0] * (1.0f / DIM);
  float var  = sB[0] * (1.0f / DIM) - mu * mu;
  float rstd = rsqrtf(var + 1e-5f);
  const int n = row & (NN - 1);
  const float* per = pe + (size_t)n * DIM;
  h16[(size_t)row * DIM + t] =
      (_Float16)((v0 - mu) * rstd * w[t] + b[t] + per[t]);
  h16[(size_t)row * DIM + t + 256] =
      (_Float16)((v1 - mu) * rstd * w[t + 256] + b[t + 256] + per[t + 256]);
}

// ---------------------------------------------------------------------------
// Weight conversion fp32 -> f16 (wq|wk|wv|wo packed)
// ---------------------------------------------------------------------------
__global__ __launch_bounds__(256) void cvt4(const float* __restrict__ a,
                                            const float* __restrict__ b,
                                            const float* __restrict__ c,
                                            const float* __restrict__ d,
                                            _Float16* __restrict__ o) {
  const int S = DIM * DIM;
  int i = blockIdx.x * 256 + threadIdx.x;
  if (i < S) {
    o[i]         = (_Float16)a[i];
    o[S + i]     = (_Float16)b[i];
    o[2 * S + i] = (_Float16)c[i];
    o[3 * S + i] = (_Float16)d[i];
  }
}

// ---------------------------------------------------------------------------
// GEMM: [8192 x 512] x [512 x 512].  Block = 8 waves, block tile 256x64,
// wave tile 32x64 (2 A-frags x 4 B-frags = 8 WMMA per 32-K step).
// B tile double-buffered in LDS (global->reg prefetch overlaps WMMA).
// ---------------------------------------------------------------------------
template <int MODE>
__global__ __launch_bounds__(256) void gemm512(const _Float16* __restrict__ A,
                                               const _Float16* __restrict__ Bw,
                                               const float* __restrict__ bias,
                                               const float* __restrict__ resid,
                                               void* __restrict__ outp,
                                               float oscale) {
  __shared__ _Float16 ldsB[2][32 * 64];
  const int tid = threadIdx.x;
  const int wave = tid >> 5, lane = tid & 31;
  const int row0 = blockIdx.x * 256 + wave * 32;
  const int col0 = blockIdx.y * 64;
  v8f acc[2][4] = {};

  const int sr = tid >> 3;        // 0..31 : K row within tile
  const int sc = (tid & 7) * 8;   // 0..56 : 8 halfs = 16 bytes

  // stage K-tile 0
  *(float4*)(ldsB[0] + sr * 64 + sc) =
      *(const float4*)(Bw + (size_t)sr * DIM + col0 + sc);

  float4 nb;
  for (int kt = 0; kt < DIM / 32; ++kt) {
    const int k0 = kt * 32, buf = kt & 1;
    __syncthreads();
    const bool more = (k0 + 32) < DIM;
    if (more)  // prefetch next B tile into registers (overlaps WMMAs below)
      nb = *(const float4*)(Bw + (size_t)(k0 + 32 + sr) * DIM + col0 + sc);
    if (more)  // keep a HW prefetch on the A stream too
      __builtin_prefetch(A + (size_t)row0 * DIM + k0 + 32, 0, 1);

    v16h a0 = load_a(A + (size_t)row0 * DIM + k0, DIM);
    v16h a1 = load_a(A + (size_t)(row0 + 16) * DIM + k0, DIM);
#pragma unroll
    for (int j = 0; j < 4; ++j) {
      v16h b = load_b(ldsB[buf] + j * 16, 64);
      acc[0][j] = wmma_f16(a0, b, acc[0][j]);
      acc[1][j] = wmma_f16(a1, b, acc[1][j]);
    }
    if (more) *(float4*)(ldsB[buf ^ 1] + sr * 64 + sc) = nb;
  }

  const int g = lane >> 4, n = lane & 15;
#pragma unroll
  for (int mi = 0; mi < 2; ++mi) {
#pragma unroll
    for (int j = 0; j < 4; ++j) {
      const int col = col0 + j * 16 + n;
#pragma unroll
      for (int r = 0; r < 8; ++r) {
        const int row = row0 + mi * 16 + r + 8 * g;
        float v = acc[mi][j][r];
        if (MODE == 0) {
          v = v * oscale + (bias ? bias[col] : 0.0f);
          const int head = col >> 6, d = col & 63;
          const int bi = row >> 11, nn = row & (NN - 1);
          ((_Float16*)outp)[((((size_t)bi * HEADS + head) * NN + nn) << 6) + d] =
              (_Float16)v;
        } else {
          ((float*)outp)[(size_t)row * DIM + col] =
              v + bias[col] + resid[(size_t)row * DIM + col];
        }
      }
    }
  }
}

// ---------------------------------------------------------------------------
// Flash attention: grid (N/128, B*H), block = 8 waves, 16 query rows / wave.
// 64-key K/V tiles moved by the Tensor Data Mover (double-buffered via
// TENSORcnt: issue tile t+1's pair, wait <=2 => tile t complete, barrier).
// Per tile: 8 WMMA for S=Q*K^T, row-max via DPP row_ror (VALU), row-sum via
// P x ones WMMA, P bounced through per-wave LDS scratch, 8 WMMA for O+=P*V.
// Q is pre-scaled by 1/sqrt(64) in the Q projection.
// ---------------------------------------------------------------------------
__global__ __launch_bounds__(256) void attn(const _Float16* __restrict__ Q,
                                            const _Float16* __restrict__ K,
                                            const _Float16* __restrict__ V,
                                            _Float16* __restrict__ O) {
  __shared__ _Float16 ldsK[2][KT * DHEAD];
  __shared__ _Float16 ldsV[2][KT * DHEAD];
  __shared__ _Float16 ldsP[8][16 * KT];
  const int tid = threadIdx.x, wave = tid >> 5, lane = tid & 31;
  const int bh = blockIdx.y;
  const int q0 = blockIdx.x * 128 + wave * 16;
  const _Float16* Qp = Q + (size_t)bh * NN * DHEAD;
  const _Float16* Kp = K + (size_t)bh * NN * DHEAD;
  const _Float16* Vp = V + (size_t)bh * NN * DHEAD;

  // kick off TDM for tile 0 before anything else
  if (tid == 0) {
    tdm_load_64x64(ldsK[0], Kp);
    tdm_load_64x64(ldsV[0], Vp);
  }

  const v16h qa0 = load_a(Qp + (size_t)q0 * DHEAD, DHEAD);
  const v16h qa1 = load_a(Qp + (size_t)q0 * DHEAD + 32, DHEAD);

  v16h ones;
#pragma unroll
  for (int i = 0; i < 16; ++i) ones[i] = (_Float16)1.0f;

  v8f o[4] = {};
  v8f lacc = {};  // running row-sum fragment (all 16 cols identical per row)
  float mrow[8];
#pragma unroll
  for (int r = 0; r < 8; ++r) mrow[r] = -1e30f;
  const int g = lane >> 4, cn = lane & 15;

  const int NT = NN / KT;
  for (int t = 0; t < NT; ++t) {
    const int buf = t & 1;
    __syncthreads();  // all waves done with previous tile's buffers
    if (tid == 0) {
      if (t + 1 < NT) {  // issue next tile, then wait for current (in-order)
        tdm_load_64x64(ldsK[buf ^ 1], Kp + (size_t)(t + 1) * KT * DHEAD);
        tdm_load_64x64(ldsV[buf ^ 1], Vp + (size_t)(t + 1) * KT * DHEAD);
        __builtin_amdgcn_s_wait_tensorcnt(2);
      } else {
        __builtin_amdgcn_s_wait_tensorcnt(0);
      }
    }
    __syncthreads();  // tile t now visible to all waves
    const _Float16* Kb = ldsK[buf];
    const _Float16* Vb = ldsV[buf];

    // S tile 16x64 = Q(16x64) * K^T(64x64)
    v8f s[4];
#pragma unroll
    for (int i = 0; i < 4; ++i) {
      v8f z = {};
      z    = wmma_f16(qa0, load_bt(Kb + i * 16 * DHEAD, DHEAD), z);
      s[i] = wmma_f16(qa1, load_bt(Kb + i * 16 * DHEAD + 32, DHEAD), z);
    }

    // online softmax: row-max on the VALU (DPP rotations)
#pragma unroll
    for (int r = 0; r < 8; ++r) {
      float m0 = fmaxf(fmaxf(s[0][r], s[1][r]), fmaxf(s[2][r], s[3][r]));
      m0 = rowmax16(m0);
      const float mnew = fmaxf(mrow[r], m0);
      const float corr = __expf(mrow[r] - mnew);
      mrow[r] = mnew;
      const float p0 = __expf(s[0][r] - mnew);
      const float p1 = __expf(s[1][r] - mnew);
      const float p2 = __expf(s[2][r] - mnew);
      const float p3 = __expf(s[3][r] - mnew);
      const int prow = (r + 8 * g) * KT + cn;
      ldsP[wave][prow]      = (_Float16)p0;
      ldsP[wave][prow + 16] = (_Float16)p1;
      ldsP[wave][prow + 32] = (_Float16)p2;
      ldsP[wave][prow + 48] = (_Float16)p3;
      o[0][r] *= corr; o[1][r] *= corr; o[2][r] *= corr; o[3][r] *= corr;
      lacc[r] *= corr;
    }

    const v16h pa0 = load_a(ldsP[wave], KT);
    const v16h pa1 = load_a(ldsP[wave] + 32, KT);

    // running row-sum: l = P x ones (replaces shuffle-add reduction)
    lacc = wmma_f16(pa0, ones, lacc);
    lacc = wmma_f16(pa1, ones, lacc);

    // O(16x64) += P(16x64) * V(64x64)
#pragma unroll
    for (int i = 0; i < 4; ++i) {
      o[i] = wmma_f16(pa0, load_b(Vb + i * 16, DHEAD), o[i]);
      o[i] = wmma_f16(pa1, load_b(Vb + 32 * DHEAD + i * 16, DHEAD), o[i]);
    }
  }

  // epilogue: normalize, scatter to [B, N, H*64] f16
  const int b = bh / HEADS, h = bh % HEADS;
#pragma unroll
  for (int r = 0; r < 8; ++r) {
    const float inv = 1.0f / lacc[r];
    const int row = q0 + r + 8 * g;
    const size_t base = ((size_t)b * NN + row) * DIM + h * DHEAD;
    O[base + cn]      = (_Float16)(o[0][r] * inv);
    O[base + 16 + cn] = (_Float16)(o[1][r] * inv);
    O[base + 32 + cn] = (_Float16)(o[2][r] * inv);
    O[base + 48 + cn] = (_Float16)(o[3][r] * inv);
  }
}

// ---------------------------------------------------------------------------
extern "C" void kernel_launch(void* const* d_in, const int* in_sizes, int n_in,
                              void* d_out, int out_size, void* d_ws,
                              size_t ws_size, hipStream_t stream) {
  (void)in_sizes; (void)n_in; (void)out_size; (void)ws_size;
  const float* x   = (const float*)d_in[0];
  const float* lnw = (const float*)d_in[1];
  const float* lnb = (const float*)d_in[2];
  const float* wq  = (const float*)d_in[3];
  const float* wk  = (const float*)d_in[4];
  const float* wv  = (const float*)d_in[5];
  const float* bv  = (const float*)d_in[6];
  const float* wo  = (const float*)d_in[7];
  const float* bo  = (const float*)d_in[8];
  const float* pe  = (const float*)d_in[9];

  char* ws = (char*)d_ws;
  const size_t HSZ = (size_t)M_TOTAL * DIM * sizeof(_Float16);  // 8 MB
  const size_t WSZ = (size_t)DIM * DIM * sizeof(_Float16);      // 0.5 MB
  _Float16* h16 = (_Float16*)ws; ws += HSZ;   // LN output; reused for attn out
  _Float16* w16 = (_Float16*)ws; ws += 4 * WSZ;
  _Float16* q16 = (_Float16*)ws; ws += HSZ;
  _Float16* k16 = (_Float16*)ws; ws += HSZ;
  _Float16* v16 = (_Float16*)ws; ws += HSZ;
  _Float16* a16 = h16;  // alias: h16 dead after QKV projections

  cvt4<<<(DIM * DIM + 255) / 256, 256, 0, stream>>>(wq, wk, wv, wo, w16);
  ln_pe<<<M_TOTAL, 256, 0, stream>>>(x, lnw, lnb, pe, h16);

  dim3 gg(M_TOTAL / 256, DIM / 64);
  const float qscale = 0.125f;  // 1/sqrt(DHEAD), folded into Q projection
  gemm512<0><<<gg, 256, 0, stream>>>(h16, w16 + 0 * DIM * DIM, nullptr, nullptr, q16, qscale);
  gemm512<0><<<gg, 256, 0, stream>>>(h16, w16 + 1 * DIM * DIM, nullptr, nullptr, k16, 1.0f);
  gemm512<0><<<gg, 256, 0, stream>>>(h16, w16 + 2 * DIM * DIM, bv, nullptr, v16, 1.0f);

  attn<<<dim3(NN / 128, BB * HEADS), 256, 0, stream>>>(q16, k16, v16, a16);

  gemm512<1><<<gg, 256, 0, stream>>>(a16, w16 + 3 * DIM * DIM, bo, x, d_out, 1.0f);
}
